// Block_50440095924362
// MI455X (gfx1250) — matrix operand order repaired and checked by reading, not compiled, not run on hardware
//
#include <hip/hip_runtime.h>
#include <math.h>

// ---------------------------------------------------------------------------
// GTU-2D block for MI455X (gfx1250, wave32, WMMA).
// All matmuls use v_wmma_f32_16x16x32_f16 with *fragment-packed* operands:
// each 16x32 A tile / 32x16 B tile is stored as 32 lanes x 16 f16 contiguous,
// so a fragment load is two b128 loads (no u16 gather / v_mov packing).
// ---------------------------------------------------------------------------

typedef _Float16 f16;
typedef __attribute__((ext_vector_type(16))) _Float16 v16h;
typedef __attribute__((ext_vector_type(8)))  float    v8f;

#define DIM   512
#define D1    1536
#define GLUD  1024
#define BATCH 16
#define HH    32
#define WW    32
#define NTOK  1024              /* HH*WW */
#define MROW  16384             /* BATCH*NTOK */
#define FN    64                /* FFT length per axis */
#define FN2   4096              /* FN*FN */

// ---------------- packed-fragment index helpers ----------------------------
// A tile (16x32, f16): lane = m + 16*bit3(kk); elem = (kk&7) + 8*(kk>>4)
// (matches ISA 7.12.2 16-bit A layout: VGPR j <-> K = 16*(j>>2)+8*hi+2*(j&3))
__device__ __forceinline__ int apack_idx(int m, int k, int kTiles) {
  int kk = k & 31;
  int tile = (m >> 4) * kTiles + (k >> 5);
  int lane = (m & 15) + (((kk >> 3) & 1) << 4);
  int elem = (kk & 7) + ((kk >> 4) << 3);
  return tile * 512 + lane * 16 + elem;
}
// B tile (32x16, f16): lane = n + 16*bit3(kk); elem = (kk&7) + 8*(kk>>4)
// (same k order as A-pack)
__device__ __forceinline__ int bpack_idx(int k, int n, int nTiles) {
  int kk = k & 31;
  int tile = (k >> 5) * nTiles + (n >> 4);
  int lane = (n & 15) + (((kk >> 3) & 1) << 4);
  return tile * 512 + lane * 16 + (kk & 7) + ((kk >> 4) << 3);
}

__device__ __forceinline__ v16h ldfrag(const f16* plane, int tile) {
  return *(const v16h*)(plane + ((size_t)tile << 9) + ((threadIdx.x & 31) << 4));
}

__device__ __forceinline__ v8f wmma16(v16h a, v16h b, v8f c) {
  return __builtin_amdgcn_wmma_f32_16x16x32_f16(false, a, false, b,
                                                (short)0, c, false, false);
}

// ---------------- 64x64 packed matmuls for the DFT --------------------------
// D = A@B (+ sign*A2@B2). A operands in A-pack (4x2 tiles), B operands in
// B-pack (2x4 tiles). 8 waves x 2 tiles. STOREB: 1 -> store D in B-pack,
// 0 -> store D in A-pack (both f16 LDS).
template <int STOREB>
__device__ __forceinline__ void mm64p(const f16* A, const f16* B,
                                      const f16* A2, const f16* B2,
                                      float sign, f16* D) {
  int w = threadIdx.x >> 5;
#pragma unroll
  for (int tt = 0; tt < 2; ++tt) {
    int t = w * 2 + tt, tr = t >> 2, tc = t & 3;
    v8f acc = {}; v8f acc2 = {};
#pragma unroll
    for (int tk = 0; tk < 2; ++tk) {
      acc = wmma16(ldfrag(A, tr * 2 + tk), ldfrag(B, tk * 4 + tc), acc);
      if (A2)
        acc2 = wmma16(ldfrag(A2, tr * 2 + tk), ldfrag(B2, tk * 4 + tc), acc2);
    }
    int l = threadIdx.x & 31, n = l & 15, hi = l >> 4;
#pragma unroll
    for (int j = 0; j < 8; ++j) {
      float v = acc[j];
      if (A2) v += sign * acc2[j];
      int m = tr * 16 + j + 8 * hi, c = tc * 16 + n;
      D[STOREB ? bpack_idx(m, c, 4) : apack_idx(m, c, 2)] = (f16)v;
    }
  }
}

// Same but stores f32 to global in A-pack *ordering* (for the coef spectrum).
__device__ __forceinline__ void mm64p_gstoreA(const f16* A, const f16* B,
                                              const f16* A2, const f16* B2,
                                              float sign, float* D) {
  int w = threadIdx.x >> 5;
#pragma unroll
  for (int tt = 0; tt < 2; ++tt) {
    int t = w * 2 + tt, tr = t >> 2, tc = t & 3;
    v8f acc = {}; v8f acc2 = {};
#pragma unroll
    for (int tk = 0; tk < 2; ++tk) {
      acc  = wmma16(ldfrag(A, tr * 2 + tk),  ldfrag(B, tk * 4 + tc),  acc);
      acc2 = wmma16(ldfrag(A2, tr * 2 + tk), ldfrag(B2, tk * 4 + tc), acc2);
    }
    int l = threadIdx.x & 31, n = l & 15, hi = l >> 4;
#pragma unroll
    for (int j = 0; j < 8; ++j) {
      int m = tr * 16 + j + 8 * hi, c = tc * 16 + n;
      D[apack_idx(m, c, 2)] = acc[j] + sign * acc2[j];
    }
  }
}

// ---------------- generic packed WMMA GEMM ----------------------------------
// C[M,N] = epi(A@B0 + bias0 [, A@B1 + bias1]) (+resid). A: A-pack, B: B-pack.
// block 256 thr (8 waves); block tile 128x64; wave tile 16x64.
// MODE: 0 none, 1 silu, 2 GLU silu(r0)*r1. PACKH: write outH in A-pack (K=N).
template <int MODE, bool HASB1, bool HASRES, bool HASF, bool HASH, bool PACKH>
__global__ __launch_bounds__(256) void gemm_kernel(
    const f16* __restrict__ A, const f16* __restrict__ B0,
    const float* __restrict__ bias0, const f16* __restrict__ B1,
    const float* __restrict__ bias1, const float* __restrict__ resid,
    float* __restrict__ outF, f16* __restrict__ outH, int M, int N, int K) {
  int w = threadIdx.x >> 5;
  int row0 = blockIdx.y * 128 + w * 16;
  int col0 = blockIdx.x * 64;
  int kT = K >> 5, nT = N >> 4;
  int tm = row0 >> 4, tn0 = col0 >> 4;
  int lane16 = (threadIdx.x & 31) << 4;
  v8f acc[4]  = {{}, {}, {}, {}};
  v8f acc2[4] = {{}, {}, {}, {}};
  for (int tk = 0; tk < kT; ++tk) {
    const f16* ap = A + (((size_t)tm * kT + tk) << 9) + lane16;
    v16h a = *(const v16h*)ap;
    if (tk + 1 < kT) __builtin_prefetch(ap + 512, 0, 1);  // global_prefetch
#pragma unroll
    for (int cc = 0; cc < 4; ++cc) {
      const f16* bp = B0 + (((size_t)tk * nT + tn0 + cc) << 9) + lane16;
      acc[cc] = wmma16(a, *(const v16h*)bp, acc[cc]);
      if (HASB1) {
        const f16* bp1 = B1 + (((size_t)tk * nT + tn0 + cc) << 9) + lane16;
        acc2[cc] = wmma16(a, *(const v16h*)bp1, acc2[cc]);
      }
    }
  }
  int l = threadIdx.x & 31, n = l & 15, hi = l >> 4;
#pragma unroll
  for (int cc = 0; cc < 4; ++cc) {
#pragma unroll
    for (int j = 0; j < 8; ++j) {
      int r  = row0 + j + 8 * hi;
      int cN = col0 + cc * 16 + n;
      float val = acc[cc][j] + bias0[cN];
      if (MODE == 1) val = val / (1.f + __expf(-val));
      if (MODE == 2) {
        float v2 = acc2[cc][j] + bias1[cN];
        val = (val / (1.f + __expf(-val))) * v2;
      }
      size_t idx = (size_t)r * N + cN;
      if (HASRES) val += resid[idx];
      if (HASF)   outF[idx] = val;
      if (HASH)
        outH[PACKH ? (size_t)apack_idx(r, cN, N >> 5) : idx] = (f16)val;
    }
  }
}

// ---------------- twiddle init (A-pack and B-pack copies) -------------------
// F[u][k] = exp(-2*pi*i*u*k/64); symmetric, so same matrix serves both sides.
__global__ void twiddle_kernel(f16* __restrict__ FAr, f16* __restrict__ FAi,
                               f16* __restrict__ FBr, f16* __restrict__ FBi) {
  int i = blockIdx.x * 256 + threadIdx.x;
  if (i < FN2) {
    int u = i >> 6, k = i & 63;
    int t = (u * k) & 63;
    float th = 6.28318530717958647692f * (float)t * (1.f / 64.f);
    f16 cr = (f16)cosf(th);
    f16 ci = (f16)(-sinf(th));
    FAr[apack_idx(u, k, 2)] = cr;
    FAi[apack_idx(u, k, 2)] = ci;
    FBr[bpack_idx(u, k, 4)] = cr;
    FBi[bpack_idx(u, k, 4)] = ci;
  }
}

// ---------------- RPE MLP (tiny, VALU) --------------------------------------
__global__ __launch_bounds__(256) void rpe_kernel(
    const float* __restrict__ pos_w, const float* __restrict__ pos_b,
    const float* __restrict__ lw, const float* __restrict__ lb,
    const float* __restrict__ ow, const float* __restrict__ ob,
    float* __restrict__ coefP) {
  int p = blockIdx.x * 256 + threadIdx.x;
  if (p >= FN2) return;
  int a = p >> 6, b = p & 63;
  float di = (a < HH) ? (float)a : (float)(a - FN);
  float dj = (b < WW) ? (float)b : (float)(b - FN);
  float cur[64], nxt[64];
#pragma unroll
  for (int k = 0; k < 64; ++k)
    cur[k] = di * pos_w[k] + dj * pos_w[64 + k] + pos_b[k];
  for (int L = 0; L < 3; ++L) {
    float ss = 0.f;
#pragma unroll
    for (int k = 0; k < 64; ++k) ss += cur[k] * cur[k];
    float inv = 1.f / (sqrtf(ss * (1.f / 64.f)) + 1e-8f);
#pragma unroll
    for (int k = 0; k < 64; ++k) {
      float t = cur[k] * inv;
      cur[k] = t / (1.f + __expf(-t));   // silu(srms(x))
    }
    const float* Wl = lw + L * 4096;
    for (int j = 0; j < 64; ++j) {
      float s = lb[L * 64 + j];
#pragma unroll
      for (int k = 0; k < 64; ++k) s += cur[k] * Wl[k * 64 + j];
      nxt[j] = s;
    }
#pragma unroll
    for (int k = 0; k < 64; ++k) cur[k] = nxt[k];
  }
  float ss = 0.f;
#pragma unroll
  for (int k = 0; k < 64; ++k) ss += cur[k] * cur[k];
  float inv = 1.f / (sqrtf(ss * (1.f / 64.f)) + 1e-8f);
#pragma unroll
  for (int k = 0; k < 64; ++k) {
    float t = cur[k] * inv;
    cur[k] = t / (1.f + __expf(-t));
  }
  float decay = __expf(-1.00050033e-3f * (fabsf(di) + fabsf(dj)));  // ln(.999)
  for (int d = 0; d < D1; ++d) {
    float s = ob[d];
#pragma unroll
    for (int k = 0; k < 64; ++k) s += cur[k] * ow[k * D1 + d];
    coefP[(size_t)d * FN2 + p] = s * decay;
  }
}

// ---------------- coef spectrum: Cf = F @ coef @ F (per channel) ------------
__global__ __launch_bounds__(256) void coef_fft_kernel(
    const float* __restrict__ coefP, const f16* __restrict__ FAr,
    const f16* __restrict__ FAi, const f16* __restrict__ FBr,
    const f16* __restrict__ FBi, float* __restrict__ coefF) {
  __shared__ __align__(64) f16 P0[FN2], P1[FN2], P2[FN2];
  int c = blockIdx.x;
  const float* cp = coefP + (size_t)c * FN2;
  for (int i = threadIdx.x; i < FN2; i += 256) {
    int m = i >> 6, k = i & 63;
    P0[apack_idx(m, k, 2)] = (f16)cp[i];
  }
  __syncthreads();
  mm64p<1>(P0, FBr, nullptr, nullptr, 0.f, P1);   // Tr = P @ Fr  (B-pack)
  mm64p<1>(P0, FBi, nullptr, nullptr, 0.f, P2);   // Ti = P @ Fi
  __syncthreads();
  float* dst = coefF + (size_t)c * (2 * FN2);
  mm64p_gstoreA(FAr, P1, FAi, P2, -1.f, dst);        // Cr = Fr Tr - Fi Ti
  mm64p_gstoreA(FAr, P2, FAi, P1, +1.f, dst + FN2);  // Ci = Fr Ti + Fi Tr
}

// ---------------- per-channel 64x64 circular conv ---------------------------
// One block per (batch, channel): pad v 32x32 -> 64x64, X = F V F,
// Z = X .* Cf, Y = real(F* Z F*)/4096, crop 32x32. Planes kept packed.
__global__ __launch_bounds__(256) void conv_kernel(
    const f16* __restrict__ vbuf,     // (MROW, D1) f16 row-major
    const float* __restrict__ coefF,  // (D1, 2, FN2) f32, A-pack order
    const f16* __restrict__ FAr, const f16* __restrict__ FAi,
    const f16* __restrict__ FBr, const f16* __restrict__ FBi,
    float* __restrict__ outbuf) {     // (MROW, D1) f32 row-major
  __shared__ __align__(64) f16 sP1[FN2], sP2[FN2], sP3[FN2], sP4[FN2];
  int b = blockIdx.x / D1;
  int c = blockIdx.x % D1;
  int tid = threadIdx.x;
  for (int i = tid; i < FN2; i += 256) sP3[i] = (f16)0.f;  // V plane (A-pack)
  __syncthreads();
  for (int i = tid; i < NTOK; i += 256) {
    int ii = i >> 5, jj = i & 31;
    sP3[apack_idx(ii, jj, 2)] = vbuf[((size_t)b * NTOK + i) * D1 + c];
  }
  __syncthreads();
  mm64p<1>(sP3, FBr, nullptr, nullptr, 0.f, sP1);  // Tr = V Fr   (B-pack)
  mm64p<1>(sP3, FBi, nullptr, nullptr, 0.f, sP2);  // Ti = V Fi
  __syncthreads();
  mm64p<0>(FAr, sP1, FAi, sP2, -1.f, sP3);         // Xr = FrTr-FiTi (A-pack)
  mm64p<0>(FAr, sP2, FAi, sP1, +1.f, sP4);         // Xi = FrTi+FiTr
  __syncthreads();
  const float* Cr = coefF + (size_t)c * (2 * FN2);
  const float* Ci = Cr + FN2;
  for (int i = tid; i < FN2; i += 256) {  // Z = X .* C, all A-pack ordered
    float xr = (float)sP3[i], xi = (float)sP4[i];
    float cr = Cr[i], ci = Ci[i];
    sP1[i] = (f16)(xr * cr - xi * ci);    // Zr (A-pack)
    sP2[i] = (f16)(xr * ci + xi * cr);    // Zi
  }
  __syncthreads();
  mm64p<1>(sP1, FBr, sP2, FBi, +1.f, sP3);  // Ur = ZrFr+ZiFi (B-pack)
  mm64p<1>(sP2, FBr, sP1, FBi, -1.f, sP4);  // Ui = ZiFr-ZrFi
  __syncthreads();
  // Yr = Fr Ur + Fi Ui ; crop tiles (rows/cols < 32), scale 1/4096
  int w = tid >> 5;
  if (w < 4) {
    int tr = w >> 1, tc = w & 1;
    v8f acc = {}; v8f acc2 = {};
#pragma unroll
    for (int tk = 0; tk < 2; ++tk) {
      acc  = wmma16(ldfrag(FAr, tr * 2 + tk), ldfrag(sP3, tk * 4 + tc), acc);
      acc2 = wmma16(ldfrag(FAi, tr * 2 + tk), ldfrag(sP4, tk * 4 + tc), acc2);
    }
    int l = tid & 31, n = l & 15, hi = l >> 4;
#pragma unroll
    for (int j = 0; j < 8; ++j) {
      int mrow = tr * 16 + j + 8 * hi;
      int col  = tc * 16 + n;
      outbuf[((size_t)b * NTOK + mrow * WW + col) * D1 + c] =
          (acc[j] + acc2[j]) * (1.f / 4096.f);
    }
  }
}

// ---------------- pack / elementwise helpers --------------------------------
__global__ void packA_kernel(const float* __restrict__ in, f16* __restrict__ o,
                             int M, int K) {
  size_t n = (size_t)M * K;
  size_t i = (size_t)blockIdx.x * blockDim.x + threadIdx.x;
  size_t st = (size_t)gridDim.x * blockDim.x;
  for (; i < n; i += st) {
    int r = (int)(i / K), k = (int)(i % K);
    o[apack_idx(r, k, K >> 5)] = (f16)in[i];
  }
}
__global__ void packB_kernel(const float* __restrict__ in, f16* __restrict__ o,
                             int K, int N) {
  size_t n = (size_t)K * N;
  size_t i = (size_t)blockIdx.x * blockDim.x + threadIdx.x;
  size_t st = (size_t)gridDim.x * blockDim.x;
  for (; i < n; i += st) {
    int k = (int)(i / N), c = (int)(i % N);
    o[bpack_idx(k, c, N >> 4)] = (f16)in[i];
  }
}
// g = u * conv, written A-packed for the o-projection (K = D1)
__global__ void gate_kernel(const f16* __restrict__ u,
                            const float* __restrict__ conv,
                            f16* __restrict__ g) {
  size_t n = (size_t)MROW * D1;
  size_t i = (size_t)blockIdx.x * blockDim.x + threadIdx.x;
  size_t st = (size_t)gridDim.x * blockDim.x;
  for (; i < n; i += st) {
    int r = (int)(i / D1), c = (int)(i % D1);
    g[apack_idx(r, c, D1 >> 5)] = (f16)((float)u[i] * conv[i]);
  }
}
// out = x1 + srms(mlp), one block per row of 512
__global__ __launch_bounds__(256) void final_kernel(
    const float* __restrict__ x1, const float* __restrict__ mlp,
    float* __restrict__ out) {
  int row = blockIdx.x;
  __shared__ float red[256];
  const float* mr = mlp + (size_t)row * DIM;
  float ss = 0.f;
  for (int i = threadIdx.x; i < DIM; i += 256) {
    float m = mr[i];
    ss += m * m;
  }
  red[threadIdx.x] = ss;
  __syncthreads();
  for (int s = 128; s > 0; s >>= 1) {
    if (threadIdx.x < s) red[threadIdx.x] += red[threadIdx.x + s];
    __syncthreads();
  }
  float inv = 1.f / (sqrtf(red[0] * (1.f / (float)DIM)) + 1e-8f);
  const float* xr = x1 + (size_t)row * DIM;
  float* o = out + (size_t)row * DIM;
  for (int i = threadIdx.x; i < DIM; i += 256) o[i] = xr[i] + mr[i] * inv;
}

// ---------------- host-side orchestration -----------------------------------
extern "C" void kernel_launch(void* const* d_in, const int* in_sizes, int n_in,
                              void* d_out, int out_size, void* d_ws,
                              size_t ws_size, hipStream_t stream) {
  (void)in_sizes; (void)n_in; (void)out_size; (void)ws_size;
  const float* x      = (const float*)d_in[0];
  const float* u_w    = (const float*)d_in[1];
  const float* u_b    = (const float*)d_in[2];
  const float* v_w    = (const float*)d_in[3];
  const float* v_b    = (const float*)d_in[4];
  const float* o_w    = (const float*)d_in[5];
  const float* o_b    = (const float*)d_in[6];
  const float* pos_w  = (const float*)d_in[7];
  const float* pos_b  = (const float*)d_in[8];
  const float* rpe_lw = (const float*)d_in[9];
  const float* rpe_lb = (const float*)d_in[10];
  const float* rpe_ow = (const float*)d_in[11];
  const float* rpe_ob = (const float*)d_in[12];
  const float* l1_w   = (const float*)d_in[13];
  const float* l1_b   = (const float*)d_in[14];
  const float* l2_w   = (const float*)d_in[15];
  const float* l2_b   = (const float*)d_in[16];
  const float* l3_w   = (const float*)d_in[17];
  const float* l3_b   = (const float*)d_in[18];
  float* out = (float*)d_out;

  char* ws = (char*)d_ws;
  size_t off = 0;
  auto alloc = [&](size_t bytes) -> void* {
    void* p = ws + off;
    off += (bytes + 255) & ~(size_t)255;
    return p;
  };
  f16* xb   = (f16*)alloc((size_t)MROW * DIM * 2);   // x A-pack; reused x1b
  f16* uwb  = (f16*)alloc((size_t)DIM * D1 * 2);     // B-pack weights
  f16* vwb  = (f16*)alloc((size_t)DIM * D1 * 2);
  f16* owb  = (f16*)alloc((size_t)D1 * DIM * 2);
  f16* l1b  = (f16*)alloc((size_t)DIM * GLUD * 2);
  f16* l2b  = (f16*)alloc((size_t)DIM * GLUD * 2);
  f16* l3b  = (f16*)alloc((size_t)GLUD * DIM * 2);
  f16* FAr  = (f16*)alloc((size_t)FN2 * 2);
  f16* FAi  = (f16*)alloc((size_t)FN2 * 2);
  f16* FBr  = (f16*)alloc((size_t)FN2 * 2);
  f16* FBi  = (f16*)alloc((size_t)FN2 * 2);
  f16* u_bf = (f16*)alloc((size_t)MROW * D1 * 2);    // row-major; reused h_bf
  f16* v_bf = (f16*)alloc((size_t)MROW * D1 * 2);    // row-major; reused g
  float* coefP    = (float*)alloc((size_t)D1 * FN2 * 4);
  float* coefF    = (float*)alloc((size_t)D1 * 2 * FN2 * 4);
  float* conv_out = (float*)alloc((size_t)MROW * D1 * 4);  // reused mlp
  float* x1       = (float*)alloc((size_t)MROW * DIM * 4);
  f16* x1b  = xb;        // xb consumed by u/v GEMMs before x1b write
  f16* g_p  = v_bf;      // v consumed by conv before gate writes
  f16* h_p  = u_bf;      // u consumed by gate before GLU writes
  float* mlp = conv_out; // conv_out consumed by gate before l3 writes

  // 1) pack inputs/weights to fragment layouts (f32 -> f16)
  packA_kernel<<<4096, 256, 0, stream>>>(x, xb, MROW, DIM);
  packB_kernel<<<768, 256, 0, stream>>>(u_w, uwb, DIM, D1);
  packB_kernel<<<768, 256, 0, stream>>>(v_w, vwb, DIM, D1);
  packB_kernel<<<768, 256, 0, stream>>>(o_w, owb, D1, DIM);
  packB_kernel<<<512, 256, 0, stream>>>(l1_w, l1b, DIM, GLUD);
  packB_kernel<<<512, 256, 0, stream>>>(l2_w, l2b, DIM, GLUD);
  packB_kernel<<<512, 256, 0, stream>>>(l3_w, l3b, GLUD, DIM);

  // 2) twiddles, RPE coefficients, coef spectrum
  twiddle_kernel<<<16, 256, 0, stream>>>(FAr, FAi, FBr, FBi);
  rpe_kernel<<<16, 256, 0, stream>>>(pos_w, pos_b, rpe_lw, rpe_lb,
                                     rpe_ow, rpe_ob, coefP);
  coef_fft_kernel<<<D1, 256, 0, stream>>>(coefP, FAr, FAi, FBr, FBi, coefF);

  // 3) u = silu(x@u_w+b), v = silu(x@v_w+b)  (row-major f16 outputs)
  gemm_kernel<1, false, false, false, true, false>
      <<<dim3(D1 / 64, MROW / 128), 256, 0, stream>>>(
          xb, uwb, u_b, nullptr, nullptr, nullptr, nullptr, u_bf,
          MROW, D1, DIM);
  gemm_kernel<1, false, false, false, true, false>
      <<<dim3(D1 / 64, MROW / 128), 256, 0, stream>>>(
          xb, vwb, v_b, nullptr, nullptr, nullptr, nullptr, v_bf,
          MROW, D1, DIM);

  // 4) per-channel circular conv via DFT matmuls
  conv_kernel<<<BATCH * D1, 256, 0, stream>>>(v_bf, coefF, FAr, FAi, FBr, FBi,
                                              conv_out);

  // 5) g = u * conv (A-packed)
  gate_kernel<<<4096, 256, 0, stream>>>(u_bf, conv_out, g_p);

  // 6) x1 = x + g @ o_w + o_b  (f32 + A-packed f16 copy)
  gemm_kernel<0, false, true, true, true, true>
      <<<dim3(DIM / 64, MROW / 128), 256, 0, stream>>>(
          g_p, owb, o_b, nullptr, nullptr, x, x1, x1b, MROW, DIM, D1);

  // 7) GLU: h = silu(x1@l1+b1) * (x1@l2+b2)  (A-packed)
  gemm_kernel<2, true, false, false, true, true>
      <<<dim3(GLUD / 64, MROW / 128), 256, 0, stream>>>(
          x1b, l1b, l1_b, l2b, l2_b, nullptr, nullptr, h_p, MROW, GLUD, DIM);

  // 8) mlp = h @ l3 + b3 (f32)
  gemm_kernel<0, false, false, true, false, false>
      <<<dim3(DIM / 64, MROW / 128), 256, 0, stream>>>(
          h_p, l3b, l3_b, nullptr, nullptr, nullptr, mlp, nullptr,
          MROW, DIM, GLUD);

  // 9) out = x1 + srms(mlp)
  final_kernel<<<MROW, 256, 0, stream>>>(x1, mlp, out);
}
